// TextPromptCrossAttention_6399501271560
// MI455X (gfx1250) — matrix operand order repaired and checked by reading, not compiled, not run on hardware
//
#include <hip/hip_runtime.h>

// ---------------------------------------------------------------------------
// TextPromptCrossAttention for MI455X (gfx1250, wave32, WMMA)
//
// HBM roofline: 613MB out + 136MB in => ~32us at 23.3 TB/s; only ~80 GFLOP of
// math => bf16 WMMA 16x16x32 everywhere. L2 weight traffic kept under fabric
// budget via 2 batches/block + weight-fragment reuse across 3 M-tiles
// (~1.9GB total, ~60 TB/s). 152KB LDS => 2 workgroups per 320KB WGP.
// K-loops are '#pragma unroll 1' to keep WMMA operands in registers
// (round-2 full unroll spilled fragments to scratch).
// ---------------------------------------------------------------------------

typedef __attribute__((ext_vector_type(16))) __bf16 v16bf;
typedef __attribute__((ext_vector_type(8)))  __bf16 v8bf;
typedef __attribute__((ext_vector_type(4)))  __bf16 v4bf;
typedef __attribute__((ext_vector_type(8)))  float  v8f;
typedef __attribute__((ext_vector_type(4)))  float  v4f;
typedef __attribute__((ext_vector_type(4)))  int    v4i;

#define BT    16
#define NTOK  77
#define TJ    17
#define DIM   512
#define NH    8
#define HD    64
#define BATCH 3888
#define RB    2        // batches fused per block
#define MROWS 48       // 2*17 KV rows padded to 3 WMMA M-tiles
#define VTS   56       // vT row stride (elements): 2 batches * 24 + slack

// cross-attn LDS: [xn 48x512 bf16 | kx 48x512 bf16 | vT 512x56 bf16]
// xn region is reused for S (f32, stride 18) then P (bf16) interleaved in-place.
#define LDS_K    49152
#define LDS_VT   98304
#define LDS_TOTAL 155648   // 152KB -> 2 blocks/WGP

// ---- WMMA helpers (ISA 7.12.2 16-bit operand layouts) ---------------------

__device__ __forceinline__ v8f wmma_bf16(v16bf a, v16bf b, v8f c) {
  return __builtin_amdgcn_wmma_f32_16x16x32_bf16(false, a, false, b,
                                                 (short)0, c, false, false);
}

// A fragment: lane m=l%16, elems 0..15 -> K=kbase+hi*16..kbase+hi*16+15
__device__ __forceinline__ v16bf load_a16(const __bf16* row, int kbase, int hi) {
  v8bf lo = *(const v8bf*)(row + kbase + hi * 16);
  v8bf hh = *(const v8bf*)(row + kbase + hi * 16 + 8);
  return __builtin_shufflevector(lo, hh, 0, 1, 2, 3, 4, 5, 6, 7,
                                         8, 9, 10, 11, 12, 13, 14, 15);
}

// A fragment from an 8-byte-aligned row (P rows live in 72B slots)
__device__ __forceinline__ v16bf load_a16_a8(const __bf16* row, int hi) {
  v4bf p0 = *(const v4bf*)(row + hi * 16);
  v4bf p1 = *(const v4bf*)(row + hi * 16 + 4);
  v4bf p2 = *(const v4bf*)(row + hi * 16 + 8);
  v4bf p3 = *(const v4bf*)(row + hi * 16 + 12);
  v8bf lo = __builtin_shufflevector(p0, p1, 0, 1, 2, 3, 4, 5, 6, 7);
  v8bf hh = __builtin_shufflevector(p2, p3, 0, 1, 2, 3, 4, 5, 6, 7);
  return __builtin_shufflevector(lo, hh, 0, 1, 2, 3, 4, 5, 6, 7,
                                         8, 9, 10, 11, 12, 13, 14, 15);
}

// B fragment: 16 contiguous bf16, 16B aligned
__device__ __forceinline__ v16bf load_b16_a16(const __bf16* p) {
  v8bf lo = *(const v8bf*)(p);
  v8bf hh = *(const v8bf*)(p + 8);
  return __builtin_shufflevector(lo, hh, 0, 1, 2, 3, 4, 5, 6, 7,
                                         8, 9, 10, 11, 12, 13, 14, 15);
}

// One wave LayerNorms one 512-element row: f32 global -> bf16 LDS
__device__ __forceinline__ void ln_row_wave(const float* __restrict__ src,
                                            const float* __restrict__ g,
                                            const float* __restrict__ bb,
                                            __bf16* dst, int lane) {
  v4f vv[4];
  float s = 0.f, s2 = 0.f;
#pragma unroll
  for (int t = 0; t < 4; ++t) {
    v4f q = *(const v4f*)(src + lane * 4 + t * 128);
    vv[t] = q;
#pragma unroll
    for (int e = 0; e < 4; ++e) { s += q[e]; s2 += q[e] * q[e]; }
  }
#pragma unroll
  for (int off = 16; off >= 1; off >>= 1) {
    s += __shfl_xor(s, off, 32);
    s2 += __shfl_xor(s2, off, 32);
  }
  float mu = s * (1.f / DIM);
  float var = s2 * (1.f / DIM) - mu * mu;
  float rs = rsqrtf(var + 1e-5f);
#pragma unroll
  for (int t = 0; t < 4; ++t) {
    int d = lane * 4 + t * 128;
    v4f g4 = *(const v4f*)(g + d);
    v4f b4 = *(const v4f*)(bb + d);
    v4bf o;
#pragma unroll
    for (int e = 0; e < 4; ++e) o[e] = (__bf16)((vv[t][e] - mu) * rs * g4[e] + b4[e]);
    *(v4bf*)(dst + d) = o;
  }
}

// ---- kernel 0: f32 -> bf16 for all three weight matrices ------------------

__global__ void convert_weights_kernel(const float* __restrict__ Wq,
                                       const float* __restrict__ Wk,
                                       const float* __restrict__ Wv,
                                       __bf16* __restrict__ WqB,
                                       __bf16* __restrict__ WkB,
                                       __bf16* __restrict__ WvB) {
  int i = blockIdx.x * 256 + threadIdx.x;   // grid covers 512*512 exactly
  WqB[i] = (__bf16)Wq[i];
  WkB[i] = (__bf16)Wk[i];
  WvB[i] = (__bf16)Wv[i];
}

// ---- kernel 1: q = LN(xf) @ Wq^T + bq (WMMA, one block per text batch) ----

__global__ void __launch_bounds__(256)
qproj_kernel(const float* __restrict__ xf, const float* __restrict__ tn_g,
             const float* __restrict__ tn_b, const __bf16* __restrict__ WqB,
             const float* __restrict__ bq, __bf16* __restrict__ qB) {
  extern __shared__ char smem[];
  __bf16* xq = (__bf16*)smem;              // 80 x 512 bf16
  const int bt = blockIdx.x;
  const int tid = threadIdx.x;
  const int w = tid >> 5, lane = tid & 31;
  const int l16 = lane & 15, hi = lane >> 4;

  for (int r = w; r < NTOK; r += 8)
    ln_row_wave(xf + (size_t)(bt * NTOK + r) * DIM, tn_g, tn_b, xq + r * DIM, lane);
  for (int idx = tid; idx < 192; idx += 256)           // zero rows 77..79
    ((v4i*)(xq + NTOK * DIM))[idx] = (v4i){0, 0, 0, 0};
  __syncthreads();

#pragma unroll 1
  for (int nt = w; nt < 32; nt += 8) {
    const int n0 = nt * 16;
    const __bf16* brow = WqB + (size_t)(n0 + l16) * DIM + hi * 16;
    v8f acc[5] = {};
#pragma unroll 1
    for (int kt = 0; kt < 16; ++kt) {
      v16bf bb = *(const v16bf*)(brow + kt * 32);
#pragma unroll
      for (int m = 0; m < 5; ++m) {
        v16bf a = load_a16(xq + (m * 16 + l16) * DIM, kt * 32, hi);
        acc[m] = wmma_bf16(a, bb, acc[m]);
      }
    }
    const float bsc = bq[n0 + l16];
#pragma unroll
    for (int m = 0; m < 5; ++m)
#pragma unroll
      for (int r = 0; r < 8; ++r) {
        int ii = m * 16 + r + 8 * hi;
        if (ii < NTOK)
          qB[(size_t)(bt * NTOK + ii) * DIM + n0 + l16] = (__bf16)(acc[m][r] + bsc);
      }
  }
}

// ---- kernel 2: fused LN + KV proj + attention, 2 batches per block --------

__global__ void __launch_bounds__(256)
cross_attn_kernel(const float* __restrict__ x, const float* __restrict__ n_g,
                  const float* __restrict__ n_b,
                  const __bf16* __restrict__ WkB, const __bf16* __restrict__ WvB,
                  const float* __restrict__ bk, const float* __restrict__ bv,
                  const __bf16* __restrict__ qB, float* __restrict__ out) {
  extern __shared__ char smem[];
  __bf16* xn = (__bf16*)smem;              // 48 x 512 (rows 34..47 zero)
  __bf16* kx = (__bf16*)(smem + LDS_K);    // 48 x 512 (rows bi*17+j)
  __bf16* vT = (__bf16*)(smem + LDS_VT);   // 512 x 56 (slot bi*24+j)
  float*  sS = (float*)smem;               // reuse xn region: (h*77+i)*18 + j

  const int b0 = blockIdx.x * RB;
  const int tid = threadIdx.x;
  const int w = tid >> 5, lane = tid & 31;
  const int l16 = lane & 15, hi = lane >> 4;

  // ---- Phase A: LayerNorm 34 rows of x -> xn bf16; zero pads + vT ----------
  for (int r = w; r < RB * TJ; r += 8)
    ln_row_wave(x + (size_t)(b0 * TJ + r) * DIM, n_g, n_b, xn + r * DIM, lane);
  for (int idx = tid; idx < 896; idx += 256)           // zero xn rows 34..47
    ((v4i*)(xn + RB * TJ * DIM))[idx] = (v4i){0, 0, 0, 0};
  for (int idx = tid; idx < 3584; idx += 256)          // zero all of vT
    ((v4i*)vT)[idx] = (v4i){0, 0, 0, 0};
  __syncthreads();

  // ---- Phase B: k/v = xn @ W^T + b; B fragment reused across 3 M-tiles -----
#pragma unroll 1
  for (int t = w; t < 64; t += 8) {
    const int proj = t >> 5;
    const int nt = t & 31;
    const __bf16* W = proj ? WvB : WkB;
    const float* bias = proj ? bv : bk;
    const int n0 = nt * 16;
    const __bf16* brow = W + (size_t)(n0 + l16) * DIM + hi * 16;
    v8f acc[3] = {};
    v16bf a[3], bb;
    bb = *(const v16bf*)brow;
#pragma unroll
    for (int m = 0; m < 3; ++m) a[m] = load_a16(xn + (m * 16 + l16) * DIM, 0, hi);
#pragma unroll 1
    for (int kt = 1; kt < 16; ++kt) {                  // software-pipelined
      __builtin_prefetch((const void*)(brow + kt * 32 + 96), 0, 3);
      v16bf bn = *(const v16bf*)(brow + kt * 32);
      v16bf an[3];
#pragma unroll
      for (int m = 0; m < 3; ++m) an[m] = load_a16(xn + (m * 16 + l16) * DIM, kt * 32, hi);
#pragma unroll
      for (int m = 0; m < 3; ++m) acc[m] = wmma_bf16(a[m], bb, acc[m]);
      bb = bn;
#pragma unroll
      for (int m = 0; m < 3; ++m) a[m] = an[m];
    }
#pragma unroll
    for (int m = 0; m < 3; ++m) acc[m] = wmma_bf16(a[m], bb, acc[m]);
    const float bsc = bias[n0 + l16];
    if (proj == 0) {
#pragma unroll
      for (int m = 0; m < 3; ++m)
#pragma unroll
        for (int r = 0; r < 8; ++r)
          kx[(m * 16 + r + 8 * hi) * DIM + n0 + l16] = (__bf16)(acc[m][r] + bsc);
    } else {
#pragma unroll
      for (int m = 0; m < 3; ++m)
#pragma unroll
        for (int r = 0; r < 8; ++r) {
          int mg = m * 16 + r + 8 * hi;
          if (mg < RB * TJ) {                          // transpose-store valid rows
            int bi2 = (mg >= TJ) ? 1 : 0;
            int j = mg - TJ * bi2;
            vT[(n0 + l16) * VTS + bi2 * 24 + j] = (__bf16)(acc[m][r] + bsc);
          }
        }
    }
  }
  __syncthreads();

  // ---- Phases C/D/E per fused batch ---------------------------------------
#pragma unroll 1
  for (int bi = 0; bi < RB; ++bi) {
    const int bt = (b0 + bi) & (BT - 1);               // jnp.tile -> b % 16
    const __bf16* qb = qB + (size_t)bt * NTOK * DIM;

    // Phase C: S = q_h @ k_h^T * 0.125
#pragma unroll 1
    for (int t = w; t < 80; t += 8) {
      const int h = t / 10, r2 = t % 10, it = r2 >> 1, nt = r2 & 1;
      int i = it * 16 + l16; if (i > NTOK - 1) i = NTOK - 1;
      int krow = bi * TJ + nt * 16 + l16; if (krow > MROWS - 1) krow = MROWS - 1;
      const __bf16* arow = qb + (size_t)i * DIM + h * HD;
      const __bf16* brow = kx + krow * DIM + h * HD + hi * 16;
      v16bf a0 = load_a16(arow, 0, hi);
      v16bf b0 = *(const v16bf*)(brow);
      v16bf a1 = load_a16(arow, 32, hi);
      v16bf b1 = *(const v16bf*)(brow + 32);
      v8f c = {};
      c = wmma_bf16(a0, b0, c);
      c = wmma_bf16(a1, b1, c);
      const int j = nt * 16 + l16;
      if (j < TJ) {
#pragma unroll
        for (int r = 0; r < 8; ++r) {
          int ii = it * 16 + r + 8 * hi;
          if (ii < NTOK) sS[(h * NTOK + ii) * 18 + j] = c[r] * 0.125f;
        }
      }
    }
    __syncthreads();

    // Phase D: softmax over j; write P (bf16, 32 slots) in place over S row
#pragma unroll 1
    for (int row = tid; row < NH * NTOK; row += 256) {
      const int h = row / NTOK, i = row % NTOK;
      float* srow = sS + (h * NTOK + i) * 18;
      float mx = -1e30f;
#pragma unroll
      for (int j = 0; j < TJ; ++j) mx = fmaxf(mx, srow[j]);
      float e[TJ];
      float sum = 0.f;
#pragma unroll
      for (int j = 0; j < TJ; ++j) { float ee = __expf(srow[j] - mx); e[j] = ee; sum += ee; }
      const float inv = 1.f / sum;
      __bf16* prow = (__bf16*)srow;
      asm volatile("" ::: "memory");                   // S reads before P writes
#pragma unroll
      for (int j = 0; j < TJ; ++j) prow[j] = (__bf16)(e[j] * inv);
#pragma unroll
      for (int j = TJ; j < 32; ++j) prow[j] = (__bf16)0.f;
    }
    __syncthreads();

    // Phase E: y_h = P @ v_h (one WMMA per 16x16 tile), stream to HBM
    float* ob = out + (size_t)(b0 + bi) * NTOK * DIM;
#pragma unroll 1
    for (int t = w; t < 160; t += 8) {
      const int h = t / 20, r2 = t % 20, it = r2 >> 2, nt = r2 & 3;
      int iidx = it * 16 + l16; if (iidx > NTOK - 1) iidx = NTOK - 1;
      const __bf16* arow = (const __bf16*)(sS + (h * NTOK + iidx) * 18);
      const __bf16* brow = vT + (h * HD + nt * 16 + l16) * VTS + bi * 24 + hi * 16;
      v16bf a = load_a16_a8(arow, hi);
      v16bf bb = load_b16_a16(brow);
      v8f c = {};
      c = wmma_bf16(a, bb, c);
      const int col = h * HD + nt * 16 + l16;
#pragma unroll
      for (int r = 0; r < 8; ++r) {
        int ii = it * 16 + r + 8 * hi;
        if (ii < NTOK) ob[(size_t)ii * DIM + col] = c[r];
      }
    }
    __syncthreads();                                   // protect S/P reuse
  }
}

// ---------------------------------------------------------------------------

extern "C" void kernel_launch(void* const* d_in, const int* in_sizes, int n_in,
                              void* d_out, int out_size, void* d_ws, size_t ws_size,
                              hipStream_t stream) {
  const float* xf   = (const float*)d_in[0];
  const float* x    = (const float*)d_in[1];
  const float* tn_g = (const float*)d_in[2];
  const float* tn_b = (const float*)d_in[3];
  const float* n_g  = (const float*)d_in[4];
  const float* n_b  = (const float*)d_in[5];
  const float* Wq   = (const float*)d_in[6];
  const float* bq   = (const float*)d_in[7];
  const float* Wk   = (const float*)d_in[8];
  const float* bk   = (const float*)d_in[9];
  const float* Wv   = (const float*)d_in[10];
  const float* bv   = (const float*)d_in[11];
  float* out = (float*)d_out;

  char* ws = (char*)d_ws;
  __bf16* WkB = (__bf16*)ws;                  // 512KB
  __bf16* WvB = (__bf16*)(ws + 524288);       // 512KB
  __bf16* WqB = (__bf16*)(ws + 1048576);      // 512KB
  __bf16* qB  = (__bf16*)(ws + 1572864);      // 16*77*512*2 B

  convert_weights_kernel<<<(DIM * DIM) / 256, 256, 0, stream>>>(Wq, Wk, Wv, WqB, WkB, WvB);
  qproj_kernel<<<BT, 256, 80 * DIM * 2, stream>>>(xf, tn_g, tn_b, WqB, bq, qB);
  cross_attn_kernel<<<BATCH / RB, 256, LDS_TOTAL, stream>>>(x, n_g, n_b, WkB, WvB,
                                                            bk, bv, qB, out);
}
